// LTFGW_4389456577036
// MI455X (gfx1250) — matrix operand order repaired and checked by reading, not compiled, not run on hardware
//
#include <hip/hip_runtime.h>

// ---------------- problem constants (match reference) ----------------
#define N_NODES 20000
#define DEG     16
#define S_DIM   17          // self + 16 neighbors
#define N_T     10
#define M_T     10
#define D_DIM   128
#define N_OUTER 5
#define N_SINK  15
#define TM      100         // N_T * M_T  (flattened template columns)
#define NP      112         // TM padded to multiple of 16 for WMMA N tiles
#define MP      32          // S_DIM padded to multiple of 16 for WMMA M tiles
#define LOGP    (-2.8332133440562162f)   // log(1/17), all nodes valid
#define PU      (1.0f/17.0f)

typedef _Float16 v16h __attribute__((ext_vector_type(16)));
typedef _Float16 v8h  __attribute__((ext_vector_type(8)));
typedef float    v8f  __attribute__((ext_vector_type(8)));

__device__ __forceinline__ float waveReduceSum(float v) {
#pragma unroll
  for (int off = 16; off > 0; off >>= 1)
    v += __shfl_xor(v, off, 32);
  return v;
}

// ---------------- precompute: q=softmax(q0), logq, ||F2||^2, cc2, F2->f16 ----
__global__ void ltfgw_pre(const float* __restrict__ tf,    // (10,10,128)
                          const float* __restrict__ tmpl,  // (10,10,10)
                          const float* __restrict__ q0,    // (10,10)
                          _Float16* __restrict__ F2h,      // (112,128) zero-padded
                          float* __restrict__ b2,          // (100) ||F2||^2
                          float* __restrict__ cc2,         // (100) sum_b C2^2 q
                          float* __restrict__ logq,        // (100)
                          float* __restrict__ qv)          // (100)
{
  const int tid = threadIdx.x;
  if (tid < N_T) {
    float r[M_T];
    float mx = -1e30f;
    for (int m = 0; m < M_T; ++m) { r[m] = q0[tid*M_T+m]; mx = fmaxf(mx, r[m]); }
    float s = 0.f;
    for (int m = 0; m < M_T; ++m) s += __expf(r[m]-mx);
    float ls = mx + __logf(s);
    for (int m = 0; m < M_T; ++m) {
      qv[tid*M_T+m]   = __expf(r[m]-ls);
      logq[tid*M_T+m] = r[m]-ls;
    }
  }
  __syncthreads();
  for (int e = tid; e < TM; e += blockDim.x) {
    float s2 = 0.f;
    for (int d = 0; d < D_DIM; ++d) { float v = tf[e*D_DIM+d]; s2 += v*v; }
    b2[e] = s2;
    int t = e / M_T, a = e - t*M_T;
    float c = 0.f;
    for (int b = 0; b < M_T; ++b) { float v = tmpl[t*100 + a*10 + b]; c += v*v*qv[t*M_T+b]; }
    cc2[e] = c;
  }
  for (int e = tid; e < NP*D_DIM; e += blockDim.x) {
    int row = e >> 7;
    F2h[e] = (row < TM) ? (_Float16)tf[e] : (_Float16)0.0f;
  }
}

// gw_tens: outG[s][t*10+v] = cc1[s] + cc2[t*10+v] - 2 * (C1 @ plan_t @ C2_t)[s][v]
__device__ __forceinline__ void gw_tens_dev(int lane,
                                            const float (*C1)[S_DIM],
                                            const float (*plan)[TM],
                                            const float* cc1, const float* cc2,
                                            const float* C2,      // LDS (10,10,10)
                                            float (*outG)[TM])
{
  for (int e = lane; e < N_T*S_DIM; e += 32) {
    int t = e / S_DIM, s = e - t*S_DIM;
    float Am[M_T];
#pragma unroll
    for (int m = 0; m < M_T; ++m) Am[m] = 0.f;
    for (int u = 0; u < S_DIM; ++u) {
      float c = C1[s][u];
#pragma unroll
      for (int m = 0; m < M_T; ++m) Am[m] += c * plan[u][t*M_T+m];
    }
    float base = cc1[s];
#pragma unroll
    for (int v = 0; v < M_T; ++v) {
      float acc = 0.f;
#pragma unroll
      for (int m = 0; m < M_T; ++m) acc += Am[m] * C2[t*100 + m*10 + v];
      outG[s][t*M_T+v] = base + cc2[t*M_T+v] - 2.0f*acc;
    }
  }
}

// ---------------- main: one wave32 per node ----------------
__global__ __launch_bounds__(32)
void ltfgw_main(const float* __restrict__ x,
                const int* __restrict__ dst,
                const float* __restrict__ tmpl,
                const float* __restrict__ alpha0,
                const _Float16* __restrict__ F2h,
                const float* __restrict__ b2g,
                const float* __restrict__ cc2g,
                const float* __restrict__ logqg,
                const float* __restrict__ qg,
                float* __restrict__ out)
{
  __shared__ _Float16 sF1[MP][D_DIM];     // padded A matrix (f16)
  __shared__ float sMc[MP][NP];           // padded so WMMA stores are branch-free
  __shared__ float sLK[S_DIM][TM];        // G / logK / final tens
  __shared__ float sPl[S_DIM][TM];        // transport plan
  __shared__ float sC1[S_DIM][S_DIM];
  __shared__ float sCc1[S_DIM];
  __shared__ float sA2[MP];               // padded ||F1_s||^2
  __shared__ float sB2[NP];               // padded ||F2_tm||^2
  __shared__ float sCc2[TM], sLogq[TM], sQ[TM];
  __shared__ float sC2[N_T*M_T*M_T];
  __shared__ float sF[N_T*S_DIM];         // f[t][s]
  __shared__ float sG[TM];                // g[t][m]
  __shared__ float sEps[N_T];
  __shared__ int   sNodes[S_DIM];
  __shared__ int   sNbr[S_DIM][DEG];

  const int n = blockIdx.x;
  const int lane = threadIdx.x;

  // constants -> LDS
  for (int e = lane; e < TM; e += 32) {
    sB2[e] = b2g[e]; sCc2[e] = cc2g[e]; sLogq[e] = logqg[e]; sQ[e] = qg[e];
  }
  if (lane < NP - TM) sB2[TM + lane] = 0.f;          // pad columns
  if (lane >= S_DIM)  sA2[lane] = 0.f;               // pad rows
  for (int e = lane; e < N_T*M_T*M_T; e += 32) sC2[e] = tmpl[e];

  if (lane == 0) sNodes[0] = n;
  if (lane >= 1 && lane < S_DIM) sNodes[lane] = dst[n*DEG + (lane-1)];
  __syncthreads();

  if (lane < S_DIM) __builtin_prefetch(&x[(size_t)sNodes[lane]*D_DIM], 0, 1);

  for (int e = lane; e < S_DIM*DEG; e += 32) {
    int s = e >> 4, j = e & 15;
    sNbr[s][j] = dst[(size_t)sNodes[s]*DEG + j];
  }
  __syncthreads();

  // gather F1 (f32->f16) and squared norms
  for (int s = 0; s < S_DIM; ++s) {
    const float* row = &x[(size_t)sNodes[s]*D_DIM];
    float part = 0.f;
#pragma unroll
    for (int i = 0; i < 4; ++i) {
      int d = lane + 32*i;
      float v = row[d];
      sF1[s][d] = (_Float16)v;
      part += v*v;
    }
    part = waveReduceSum(part);
    if (lane == 0) sA2[s] = part;
  }
  for (int s = S_DIM; s < MP; ++s) {
#pragma unroll
    for (int i = 0; i < 4; ++i) sF1[s][lane + 32*i] = (_Float16)0.0f;
  }

  // C1 adjacency among the 17 ego-graph nodes (symmetrized)
  for (int e = lane; e < S_DIM*S_DIM; e += 32) {
    int s = e / S_DIM, u = e - s*S_DIM;
    bool m = false;
#pragma unroll
    for (int j = 0; j < DEG; ++j)
      m = m || (sNbr[s][j] == sNodes[u]) || (sNbr[u][j] == sNodes[s]);
    sC1[s][u] = m ? 1.0f : 0.0f;
  }
  __syncthreads();
  if (lane < S_DIM) {
    float r = 0.f;
    for (int u = 0; u < S_DIM; ++u) r += sC1[lane][u];
    sCc1[lane] = r * PU;                 // sum_b C1^2 * p  (C1 is 0/1)
  }
  __syncthreads();

  // ---- WMMA GEMM: Mc[s][tm] = ||F1_s||^2 + ||F2_tm||^2 - 2 F1.F2 ----
  // A = sF1 (32x128 f16 LDS), B = F2h (112x128 f16, used K-major), f32 accum.
  // Store into padded sMc[32][112] -> completely branch-free epilogue.
  {
    const int l15 = lane & 15;
    const int h   = lane >> 4;
    for (int mt = 0; mt < 2; ++mt) {
      const _Float16* arow = &sF1[mt*16 + l15][0];
      for (int nt = 0; nt < 7; ++nt) {
        v8f acc = {0.f,0.f,0.f,0.f,0.f,0.f,0.f,0.f};
        const _Float16* brow = F2h + (size_t)(nt*16 + l15)*D_DIM;
#pragma unroll
        for (int kc = 0; kc < 4; ++kc) {
          // A fragment: lane row = mt*16+l15, half h holds K = kc*32 + h*16 .. +15
          int kbA = kc*32 + h*16;
          v8h alo = *(const v8h*)(arow + kbA);
          v8h ahi = *(const v8h*)(arow + kbA + 8);
          v16h a = __builtin_shufflevector(alo, ahi,
                     0,1,2,3,4,5,6,7,8,9,10,11,12,13,14,15);
          // B fragment: lane col = nt*16+l15, half h holds K = kc*32 + h*16 .. +15
          v16h b = *(const v16h*)(brow + kc*32 + h*16);
          acc = __builtin_amdgcn_wmma_f32_16x16x32_f16(
                  false, a, false, b, (short)0, acc, false, false);
        }
        const int ncol = nt*16 + l15;
#pragma unroll
        for (int i = 0; i < 8; ++i) {
          int srow = mt*16 + h*8 + i;        // documented f32 C/D layout
          sMc[srow][ncol] = sA2[srow] + sB2[ncol] - 2.0f*acc[i];
        }
      }
    }
  }
  __syncthreads();

  const float al    = 1.0f/(1.0f + __expf(-alpha0[n]));
  const float one_m = 1.0f - al;

  // plan0 = p (x) q
  for (int e = lane; e < S_DIM*TM; e += 32) {
    int s = e / TM, tm = e - s*TM;
    sPl[s][tm] = PU * sQ[tm];
  }
  __syncthreads();

  for (int it = 0; it < N_OUTER; ++it) {
    gw_tens_dev(lane, sC1, sPl, sCc1, sCc2, sC2, sLK);
    __syncthreads();
    // G = (1-a)*Mc + 2a*tens
    for (int e = lane; e < S_DIM*TM; e += 32) {
      int s = e / TM, tm = e - s*TM;
      sLK[s][tm] = one_m*sMc[s][tm] + 2.0f*al*sLK[s][tm];
    }
    __syncthreads();
    if (lane < N_T) {
      float s = 0.f;
      for (int ss = 0; ss < S_DIM; ++ss)
        for (int m = 0; m < M_T; ++m) s += fabsf(sLK[ss][lane*M_T+m]);
      sEps[lane] = 0.05f*(s/(float)(S_DIM*M_T)) + 1e-8f;
    }
    __syncthreads();
    // logK = -G/eps ; g = 0
    for (int e = lane; e < S_DIM*TM; e += 32) {
      int s = e / TM, tm = e - s*TM;
      sLK[s][tm] = -sLK[s][tm] / sEps[tm/M_T];
    }
    for (int e = lane; e < TM; e += 32) sG[e] = 0.f;
    __syncthreads();

    for (int si = 0; si <= N_SINK; ++si) {
      // f[t][s] = logp - LSE_m(logK + g)
      for (int e = lane; e < N_T*S_DIM; e += 32) {
        int t = e / S_DIM, s = e - t*S_DIM;
        float mx = -1e30f;
#pragma unroll
        for (int m = 0; m < M_T; ++m) mx = fmaxf(mx, sLK[s][t*M_T+m] + sG[t*M_T+m]);
        float sum = 0.f;
#pragma unroll
        for (int m = 0; m < M_T; ++m) sum += __expf(sLK[s][t*M_T+m] + sG[t*M_T+m] - mx);
        sF[e] = LOGP - (mx + __logf(sum));
      }
      __syncthreads();
      if (si == N_SINK) break;
      // g[t][m] = logq - LSE_s(logK + f)
      for (int e = lane; e < TM; e += 32) {
        int t = e / M_T;
        float mx = -1e30f;
        for (int s = 0; s < S_DIM; ++s) mx = fmaxf(mx, sLK[s][e] + sF[t*S_DIM+s]);
        float sum = 0.f;
        for (int s = 0; s < S_DIM; ++s) sum += __expf(sLK[s][e] + sF[t*S_DIM+s] - mx);
        sG[e] = sLogq[e] - (mx + __logf(sum));
      }
      __syncthreads();
    }
    // plan = exp(logK + f + g)
    for (int e = lane; e < S_DIM*TM; e += 32) {
      int s = e / TM, tm = e - s*TM, t = tm / M_T;
      sPl[s][tm] = __expf(sLK[s][tm] + sF[t*S_DIM+s] + sG[tm]);
    }
    __syncthreads();
  }

  // final: tens, lin, gw
  gw_tens_dev(lane, sC1, sPl, sCc1, sCc2, sC2, sLK);
  __syncthreads();
  if (lane < N_T) {
    float lin = 0.f, gw = 0.f;
    for (int s = 0; s < S_DIM; ++s) {
      for (int m = 0; m < M_T; ++m) {
        int tm = lane*M_T + m;
        lin += sMc[s][tm] * sPl[s][tm];
        gw  += sLK[s][tm] * sPl[s][tm];
      }
    }
    out[(size_t)n*N_T + lane] = one_m*lin + al*gw;
  }
}

extern "C" void kernel_launch(void* const* d_in, const int* in_sizes, int n_in,
                              void* d_out, int out_size, void* d_ws, size_t ws_size,
                              hipStream_t stream)
{
  (void)in_sizes; (void)n_in; (void)out_size; (void)ws_size;
  const float* x      = (const float*)d_in[0];
  const int*   ei     = (const int*)d_in[1];
  const float* tmpl   = (const float*)d_in[2];
  const float* tf     = (const float*)d_in[3];
  const float* q0     = (const float*)d_in[4];
  const float* alpha0 = (const float*)d_in[5];
  float* out = (float*)d_out;

  // src = repeat(arange(N),16) -> neighbor table is dst.reshape(N,16)
  const int* dst = ei + (size_t)N_NODES*DEG;

  char* ws = (char*)d_ws;
  _Float16* F2h = (_Float16*)ws;                       // 112*128*2 = 28672 B
  float* b2   = (float*)(ws + (size_t)NP*D_DIM*2);
  float* cc2  = b2 + TM;
  float* logq = cc2 + TM;
  float* qv   = logq + TM;

  ltfgw_pre<<<1, 128, 0, stream>>>(tf, tmpl, q0, F2h, b2, cc2, logq, qv);
  ltfgw_main<<<N_NODES, 32, 0, stream>>>(x, dst, tmpl, alpha0,
                                         F2h, b2, cc2, logq, qv, out);
}
